// PoolNet_83099027243705
// MI455X (gfx1250) — matrix operand order repaired and checked by reading, not compiled, not run on hardware
//
#include <hip/hip_runtime.h>

typedef __attribute__((ext_vector_type(16))) __bf16 v16bf;
typedef __attribute__((ext_vector_type(8)))  float  v8f;
typedef unsigned int u32;

union FragU { u32 u[8]; uint4 q[2]; v16bf v; };

#define F_  32
#define P_  64
#define H_  128
#define E_  64
#define K1  192
#define M_  512
#define D_  1024
#define KB1 6      // K1/32
#define KB2 16     // M/32
#define EPS 1e-5f

__device__ __forceinline__ unsigned short f2bf(float f) {
    unsigned int u = __float_as_uint(f);
    unsigned int r = u + 0x7FFFu + ((u >> 16) & 1u);   // RNE
    return (unsigned short)(r >> 16);
}
__device__ __forceinline__ u32 pack2bf(float a, float b) {
    return (u32)f2bf(a) | ((u32)f2bf(b) << 16);
}
// WMMA A-operand map: even k_local (0..31) -> (K-half hi, dword p)
__device__ __forceinline__ void amap(int kl, int& hi, int& p) {
    hi = kl >> 4;
    p  = (kl & 15) >> 1;
}

// ---- prep: weights -> bf16 pre-swizzled into per-lane WMMA B-fragment order ----
// layout: dword index = ((nt*KB + ks)*32 + lane)*8 + p ; lane: n=lane&15, khalf=lane>>4 ;
//         dword p holds k = ks*32 + 16*khalf + 2p and k+1
__global__ __launch_bounds__(256) void poolnet_prep(
    const float* __restrict__ W1, const float* __restrict__ W2,
    const float* __restrict__ b1, const float* __restrict__ g1, const float* __restrict__ beta1,
    const float* __restrict__ b2, const float* __restrict__ g2, const float* __restrict__ beta2,
    const float* __restrict__ rm1, const float* __restrict__ rv1,
    const float* __restrict__ rm2, const float* __restrict__ rv2,
    u32* __restrict__ w1s, u32* __restrict__ w2s,
    float* __restrict__ s1, float* __restrict__ c1,
    float* __restrict__ s2, float* __restrict__ c2)
{
    int gid = blockIdx.x * 256 + threadIdx.x;   // 0 .. 262143
    {   // W2: 512x1024 -> 262144 dwords
        int p = gid & 7, L = (gid >> 3) & 31, t = gid >> 8;
        int ks = t & 15, nt = t >> 4;
        int n = nt * 16 + (L & 15);
        int k = ks * 32 + 16 * (L >> 4) + 2 * p;
        w2s[gid] = pack2bf(W2[k * D_ + n], W2[(k + 1) * D_ + n]);
    }
    if (gid < K1 * M_ / 2) {   // W1: 192x512 -> 49152 dwords
        int p = gid & 7, L = (gid >> 3) & 31, t = gid >> 8;
        int ks = t % KB1, nt = t / KB1;
        int n = nt * 16 + (L & 15);
        int k = ks * 32 + 16 * (L >> 4) + 2 * p;
        w1s[gid] = pack2bf(W1[k * M_ + n], W1[(k + 1) * M_ + n]);
    }
    if (gid < M_) {
        float s = g1[gid] * rsqrtf(rv1[gid] + EPS);
        s1[gid] = s;
        c1[gid] = (b1[gid] - rm1[gid]) * s + beta1[gid];
    } else if (gid < M_ + D_) {
        int n = gid - M_;
        float s = g2[n] * rsqrtf(rv2[n] + EPS);
        s2[n] = s;
        c2[n] = (b2[n] - rm2[n]) * s + beta2[n];
    }
}

// ---- fused pool kernel: one block per (frame f, ped i) ----
__global__ __launch_bounds__(256) void poolnet_fused(
    const float* __restrict__ hst,   // (B, H)
    const float* __restrict__ pos,   // (B, 2)
    const float* __restrict__ We,    // (2, E)
    const float* __restrict__ be,    // (E)
    const u32* __restrict__ w1s,     // swizzled bf16 W1 frags
    const u32* __restrict__ w2s,     // swizzled bf16 W2 frags
    const float* __restrict__ s1, const float* __restrict__ c1,
    const float* __restrict__ s2, const float* __restrict__ c2,
    float* __restrict__ out)         // (B, D)
{
    // A operands pre-swizzled into WMMA A-fragment layout:
    // dword = ((ks*4 + mt)*32 + lane)*8 + p
    __shared__ __align__(16) u32 A1s[KB1 * 4 * 32 * 8];   //  6144 dw = 24 KB
    __shared__ __align__(16) u32 Hds[KB2 * 4 * 32 * 8];   // 16384 dw = 64 KB

    const int blk  = blockIdx.x;
    const int f    = blk >> 6;
    const int i    = blk & 63;
    const int tid  = threadIdx.x;
    const int base = f * P_;

    const float pix = pos[2 * (base + i) + 0];
    const float piy = pos[2 * (base + i) + 1];

    // --- stage A = [ (pos_j - pos_i) @ We + be , h_j ] directly in frag layout ---
    for (int idx = tid; idx < P_ * (E_ / 2); idx += 256) {   // emb: cols 0..63
        int j = idx >> 5, col = (idx & 31) * 2;
        float dx = pos[2 * (base + j) + 0] - pix;
        float dy = pos[2 * (base + j) + 1] - piy;
        float v0 = fmaf(dx, We[col],     fmaf(dy, We[E_ + col],     be[col]));
        float v1 = fmaf(dx, We[col + 1], fmaf(dy, We[E_ + col + 1], be[col + 1]));
        int ks = col >> 5, kl = col & 31, hi, p;
        amap(kl, hi, p);
        int L = (j & 15) + 16 * hi;
        A1s[((ks * 4 + (j >> 4)) * 32 + L) * 8 + p] = pack2bf(v0, v1);
    }
    for (int idx = tid; idx < P_ * (H_ / 2); idx += 256) {   // h: cols 64..191
        int j = idx >> 6, c = (idx & 63) * 2, col = E_ + c;
        const float* hp = &hst[(base + j) * H_ + c];
        int ks = col >> 5, kl = col & 31, hi, p;
        amap(kl, hi, p);
        int L = (j & 15) + 16 * hi;
        A1s[((ks * 4 + (j >> 4)) * 32 + L) * 8 + p] = pack2bf(hp[0], hp[1]);
    }
    __syncthreads();

    const int wave = tid >> 5;
    const int lane = tid & 31;
    const int hl   = lane >> 4;
    const int ln   = lane & 15;

    // ---------------- GEMM1: (64 x 192) @ (192 x 512) -> Hds (swizzled bf16) ----------------
    for (int pass = 0; pass < 2; ++pass) {
        const int nt0 = wave * 4 + pass * 2;
        v8f acc[2][4];
        #pragma unroll
        for (int t = 0; t < 2; ++t)
            #pragma unroll
            for (int mt = 0; mt < 4; ++mt) acc[t][mt] = (v8f){};
        for (int ks = 0; ks < KB1; ++ks) {
            FragU b0, b1;
            const uint4* pb0 = (const uint4*)&w1s[(((nt0 + 0) * KB1 + ks) * 32 + lane) * 8];
            const uint4* pb1 = (const uint4*)&w1s[(((nt0 + 1) * KB1 + ks) * 32 + lane) * 8];
            b0.q[0] = pb0[0]; b0.q[1] = pb0[1];
            b1.q[0] = pb1[0]; b1.q[1] = pb1[1];
            #pragma unroll
            for (int mt = 0; mt < 4; ++mt) {
                FragU af;
                const uint4* pa = (const uint4*)&A1s[((ks * 4 + mt) * 32 + lane) * 8];
                af.q[0] = pa[0]; af.q[1] = pa[1];
                acc[0][mt] = __builtin_amdgcn_wmma_f32_16x16x32_bf16(
                    false, af.v, false, b0.v, (short)0, acc[0][mt], false, false);
                acc[1][mt] = __builtin_amdgcn_wmma_f32_16x16x32_bf16(
                    false, af.v, false, b1.v, (short)0, acc[1][mt], false, false);
            }
        }
        // epilogue: BN+ReLU, pack bf16 pairs along N via shfl, scatter into GEMM2 A layout
        #pragma unroll
        for (int t = 0; t < 2; ++t) {
            const int ntA = nt0 + t;
            const int n   = ntA * 16 + ln;
            const float sc = s1[n];
            const float bi = c1[n];
            #pragma unroll
            for (int mt = 0; mt < 4; ++mt) {
                #pragma unroll
                for (int r = 0; r < 8; ++r) {
                    float y = fmaf(acc[t][mt][r], sc, bi);
                    y = y > 0.f ? y : 0.f;
                    float yo = __shfl_xor(y, 1, 32);   // partner column n^1
                    if ((ln & 1) == 0) {
                        int m  = mt * 16 + r + 8 * hl;  // GEMM2 A row
                        int kg = n;                     // GEMM2 A col (even)
                        int ks2 = kg >> 5, kl = kg & 31, hi2, p2;
                        amap(kl, hi2, p2);
                        int L2 = (m & 15) + 16 * hi2;
                        Hds[((ks2 * 4 + (m >> 4)) * 32 + L2) * 8 + p2] = pack2bf(y, yo);
                    }
                }
            }
        }
    }
    __syncthreads();

    // ---------------- GEMM2: (64 x 512) @ (512 x 1024), BN+ReLU, max over rows ----------------
    for (int pass = 0; pass < 4; ++pass) {
        const int nt0 = wave * 8 + pass * 2;
        v8f acc[2][4];
        #pragma unroll
        for (int t = 0; t < 2; ++t)
            #pragma unroll
            for (int mt = 0; mt < 4; ++mt) acc[t][mt] = (v8f){};
        for (int ks = 0; ks < KB2; ++ks) {
            FragU b0, b1;
            const uint4* pb0 = (const uint4*)&w2s[(((nt0 + 0) * KB2 + ks) * 32 + lane) * 8];
            const uint4* pb1 = (const uint4*)&w2s[(((nt0 + 1) * KB2 + ks) * 32 + lane) * 8];
            b0.q[0] = pb0[0]; b0.q[1] = pb0[1];
            b1.q[0] = pb1[0]; b1.q[1] = pb1[1];
            #pragma unroll
            for (int mt = 0; mt < 4; ++mt) {
                FragU af;
                const uint4* pa = (const uint4*)&Hds[((ks * 4 + mt) * 32 + lane) * 8];
                af.q[0] = pa[0]; af.q[1] = pa[1];
                acc[0][mt] = __builtin_amdgcn_wmma_f32_16x16x32_bf16(
                    false, af.v, false, b0.v, (short)0, acc[0][mt], false, false);
                acc[1][mt] = __builtin_amdgcn_wmma_f32_16x16x32_bf16(
                    false, af.v, false, b1.v, (short)0, acc[1][mt], false, false);
            }
        }
        #pragma unroll
        for (int t = 0; t < 2; ++t) {
            const int n = (nt0 + t) * 16 + ln;
            const float sc = s2[n];
            const float bi = c2[n];
            float m = 0.f;   // ReLU lower-bounds the max
            #pragma unroll
            for (int mt = 0; mt < 4; ++mt) {
                #pragma unroll
                for (int r = 0; r < 8; ++r) {
                    float y = fmaf(acc[t][mt][r], sc, bi);
                    y = y > 0.f ? y : 0.f;
                    m = fmaxf(m, y);
                }
            }
            m = fmaxf(m, __shfl_xor(m, 16, 32));   // combine row-halves (wave32)
            if (lane < 16) out[blk * D_ + n] = m;
        }
    }
}

extern "C" void kernel_launch(void* const* d_in, const int* in_sizes, int n_in,
                              void* d_out, int out_size, void* d_ws, size_t ws_size,
                              hipStream_t stream) {
    (void)in_sizes; (void)n_in; (void)out_size; (void)ws_size;
    const float* hst   = (const float*)d_in[0];
    const float* pos   = (const float*)d_in[1];
    // d_in[2] = seq_start_end (unused: frames are uniform P=64)
    const float* We    = (const float*)d_in[3];
    const float* be    = (const float*)d_in[4];
    const float* W1    = (const float*)d_in[5];
    const float* b1    = (const float*)d_in[6];
    const float* g1    = (const float*)d_in[7];
    const float* beta1 = (const float*)d_in[8];
    const float* W2    = (const float*)d_in[9];
    const float* b2    = (const float*)d_in[10];
    const float* g2    = (const float*)d_in[11];
    const float* beta2 = (const float*)d_in[12];
    const float* rm1   = (const float*)d_in[13];
    const float* rv1   = (const float*)d_in[14];
    const float* rm2   = (const float*)d_in[15];
    const float* rv2   = (const float*)d_in[16];

    u32* w1s = (u32*)d_ws;                 // 49152 dwords
    u32* w2s = w1s + K1 * M_ / 2;          // 262144 dwords
    float* s1 = (float*)(w2s + M_ * D_ / 2);
    float* c1 = s1 + M_;
    float* s2 = c1 + M_;
    float* c2 = s2 + D_;

    poolnet_prep<<<(M_ * D_ / 2) / 256, 256, 0, stream>>>(
        W1, W2, b1, g1, beta1, b2, g2, beta2, rm1, rv1, rm2, rv2,
        w1s, w2s, s1, c1, s2, c2);

    poolnet_fused<<<F_ * P_, 256, 0, stream>>>(
        hst, pos, We, be, w1s, w2s, s1, c1, s2, c2, (float*)d_out);
}
